// GPT2LMHeadModel_21397527069212
// MI455X (gfx1250) — compile-verified, matrix operand order checked by reading, not run on hardware
//
#include <hip/hip_runtime.h>

// ---------------------------------------------------------------------------
// GPT-2 small forward for gfx1250 (MI455X). bf16 WMMA GEMMs, f32 accumulate.
// Double-buffered LDS pipeline: global prefetch overlaps WMMA, 1 barrier/iter.
// ---------------------------------------------------------------------------

typedef __bf16 bf16;
typedef __attribute__((ext_vector_type(16))) __bf16 v16bf;
typedef __attribute__((ext_vector_type(8)))  __bf16 bf16x8;
typedef __attribute__((ext_vector_type(8)))  float  v8f;

#define BM 128
#define BN 64
#define BK 32
#define LDSS 40   // LDS row stride in elems: 80 bytes = multiple of 16B for b128 ops

__device__ __forceinline__ float gelu_tanh(float x) {
    const float c = 0.7978845608028654f; // sqrt(2/pi)
    float t = c * (x + 0.044715f * x * x * x);
    return 0.5f * x * (1.0f + tanhf(t));
}

// Build a 16-elem bf16 fragment from two contiguous 8-elem runs (ISA 16-bit
// A/B 16x32 layout: per lane, K = base+0..7 then K = 16+base+0..7).
__device__ __forceinline__ v16bf load_frag(const bf16* p) {
    bf16x8 lo = *(const bf16x8*)p;
    bf16x8 hi = *(const bf16x8*)(p + 16);
    return __builtin_shufflevector(lo, hi, 0,1,2,3,4,5,6,7,8,9,10,11,12,13,14,15);
}

// ---------------------------------------------------------------------------
// Tiled WMMA GEMM:  C[z] = act(alpha * A[z] @ B[z] + bias) + resid
//   A: [M,K] bf16 row-major (lda), batch stride batchA
//   B: [K,N] bf16 row-major (ldb), batch stride batchB; ldb columns are fully
//      loadable (padded) so B staging needs no bounds checks.
//   C offset for batch z: (z/HH)*cstride_b + (z%HH)*cstride_h
//   ACT: 0 none, 1 gelu.  BF16OUT selects bf16 (Cb) vs f32 (Cf) output.
//   Block tile 128x64, 8 waves, 32x32 per wave => 4 WMMAs / K-step / wave.
//   Double-buffered LDS; register prefetch of the next K-tile.
// ---------------------------------------------------------------------------
template <int ACT, bool BF16OUT>
__global__ __launch_bounds__(256) void wmma_gemm(
    const bf16* __restrict__ A, long lda, long batchA,
    const bf16* __restrict__ Bw, long ldb, long batchB,
    float* __restrict__ Cf, bf16* __restrict__ Cb,
    long ldc, long cstride_b, long cstride_h, int HH,
    const float* __restrict__ bias, const float* __restrict__ resid,
    float alpha, int M, int N, int K)
{
    __shared__ bf16 As[2][BM * LDSS];
    __shared__ bf16 Bt[2][BN * LDSS];   // B tile stored transposed: Bt[col][k]

    const int tid = threadIdx.x;
    const int z = blockIdx.z;
    A  += (long)z * batchA;
    Bw += (long)z * batchB;
    const long coff = (long)(z / HH) * cstride_b + (long)(z % HH) * cstride_h;
    if (BF16OUT) Cb += coff; else Cf += coff;

    const long blockM = (long)blockIdx.y * BM;
    const long blockN = (long)blockIdx.x * BN;

    const int wave = tid >> 5;
    const int lane = tid & 31;
    const int half = lane >> 4;          // 0: K base 0   1: K base 8
    const int lrow = lane & 15;
    const int mBase = (wave & 3) * 32;   // 4 M-groups of 32 rows
    const int nBase = (wave >> 2) * 32;  // 2 N-groups of 32 cols

    // cooperative tile-load indices
    const int arow = tid >> 1;           // 0..127
    const int acol = (tid & 1) * 16;     // 0 or 16
    const int brow = tid >> 3;           // 0..31
    const int bcol = (tid & 7) * 8;      // 0..56

    const bf16* aptr = A + (blockM + arow) * lda + acol;
    const bf16* bptr = Bw + (long)brow * ldb + blockN + bcol;

    v8f acc00 = {}, acc01 = {}, acc10 = {}, acc11 = {};

    // ---- prologue: stage tile 0 into buffer 0 ----
    {
        bf16x8 ra0 = *(const bf16x8*)aptr;
        bf16x8 ra1 = *(const bf16x8*)(aptr + 8);
        bf16x8 rb  = *(const bf16x8*)bptr;
        *(bf16x8*)&As[0][arow * LDSS + acol]     = ra0;
        *(bf16x8*)&As[0][arow * LDSS + acol + 8] = ra1;
#pragma unroll
        for (int j = 0; j < 8; ++j)
            Bt[0][(bcol + j) * LDSS + brow] = rb[j];
    }
    __syncthreads();

    int p = 0;
    for (long k0 = 0; k0 < K; k0 += BK, p ^= 1) {
        const bool hasNext = (k0 + BK) < K;

        // 1) issue global loads for the NEXT tile (latency hidden by WMMAs)
        bf16x8 ra0, ra1, rb;
        if (hasNext) {
            const bf16* an = aptr + (k0 + BK);
            const bf16* bn = bptr + (k0 + BK) * ldb;
            ra0 = *(const bf16x8*)an;
            ra1 = *(const bf16x8*)(an + 8);
            rb  = *(const bf16x8*)bn;
        }

        // 2) fragments from current buffer + 4 WMMAs
        const int fo = half * 8;
        v16bf a0 = load_frag(&As[p][(mBase + lrow) * LDSS + fo]);
        v16bf a1 = load_frag(&As[p][(mBase + 16 + lrow) * LDSS + fo]);
        v16bf b0 = load_frag(&Bt[p][(nBase + lrow) * LDSS + fo]);
        v16bf b1 = load_frag(&Bt[p][(nBase + 16 + lrow) * LDSS + fo]);

        acc00 = __builtin_amdgcn_wmma_f32_16x16x32_bf16(false, a0, false, b0,
                                                        (short)0, acc00, false, false);
        acc01 = __builtin_amdgcn_wmma_f32_16x16x32_bf16(false, a0, false, b1,
                                                        (short)0, acc01, false, false);
        acc10 = __builtin_amdgcn_wmma_f32_16x16x32_bf16(false, a1, false, b0,
                                                        (short)0, acc10, false, false);
        acc11 = __builtin_amdgcn_wmma_f32_16x16x32_bf16(false, a1, false, b1,
                                                        (short)0, acc11, false, false);

        // 3) write the prefetched tile into the other buffer
        if (hasNext) {
            const int q = p ^ 1;
            *(bf16x8*)&As[q][arow * LDSS + acol]     = ra0;
            *(bf16x8*)&As[q][arow * LDSS + acol + 8] = ra1;
#pragma unroll
            for (int j = 0; j < 8; ++j)
                Bt[q][(bcol + j) * LDSS + brow] = rb[j];
        }

        // 4) single barrier: next buffer published, current buffer drained
        __syncthreads();
    }

    // ---- epilogue: C/D layout = lane->N, vgpr r -> M=r (+8 for lanes 16..31)
#pragma unroll
    for (int im = 0; im < 2; ++im) {
#pragma unroll
        for (int r = 0; r < 8; ++r) {
            const long m = blockM + mBase + im * 16 + half * 8 + r;
#pragma unroll
            for (int in = 0; in < 2; ++in) {
                const long n = blockN + nBase + in * 16 + lrow;
                if (n < N) {
                    float v;
                    if (im == 0) v = (in == 0) ? acc00[r] : acc01[r];
                    else         v = (in == 0) ? acc10[r] : acc11[r];
                    v *= alpha;
                    if (bias) v += bias[n];
                    if (ACT == 1) v = gelu_tanh(v);
                    if (resid) v += resid[m * ldc + n];
                    if (BF16OUT) Cb[m * ldc + n] = (bf16)v;
                    else         Cf[m * ldc + n] = v;
                }
            }
        }
    }
}

// ---------------------------------------------------------------------------
// Embedding: x[m,e] = wte[ids[m],e] + wpe[m%S, e]
// ---------------------------------------------------------------------------
__global__ __launch_bounds__(256) void embed_kernel(
    const int* __restrict__ ids, const float* __restrict__ wte,
    const float* __restrict__ wpe, float* __restrict__ x, int S, int E)
{
    long idx = (long)blockIdx.x * 256 + threadIdx.x;
    long m = idx / E;
    int  e = (int)(idx % E);
    int tok = ids[m];
    x[idx] = wte[(long)tok * E + e] + wpe[(long)(m % S) * E + e];
}

// ---------------------------------------------------------------------------
// LayerNorm (eps 1e-6) f32 in -> bf16 out. One block per row.
// ---------------------------------------------------------------------------
__global__ __launch_bounds__(256) void layernorm_bf16(
    const float* __restrict__ x, const float* __restrict__ s,
    const float* __restrict__ b, bf16* __restrict__ out, int E)
{
    __shared__ float red[256];
    const int tid = threadIdx.x;
    const float* xr = x + (long)blockIdx.x * E;

    float sum = 0.f;
    for (int i = tid; i < E; i += 256) sum += xr[i];
    red[tid] = sum; __syncthreads();
    for (int off = 128; off > 0; off >>= 1) {
        if (tid < off) red[tid] += red[tid + off];
        __syncthreads();
    }
    const float mu = red[0] / (float)E;
    __syncthreads();

    float vs = 0.f;
    for (int i = tid; i < E; i += 256) { float d = xr[i] - mu; vs += d * d; }
    red[tid] = vs; __syncthreads();
    for (int off = 128; off > 0; off >>= 1) {
        if (tid < off) red[tid] += red[tid + off];
        __syncthreads();
    }
    const float inv = rsqrtf(red[0] / (float)E + 1e-6f);

    bf16* orow = out + (long)blockIdx.x * E;
    for (int i = tid; i < E; i += 256)
        orow[i] = (bf16)((xr[i] - mu) * inv * s[i] + b[i]);
}

// ---------------------------------------------------------------------------
// Split QKV f32 [M,3E] -> bf16  Q[B,H,S,D], Kt[B,H,D,S] (transposed), V[B,H,S,D]
// ---------------------------------------------------------------------------
__global__ __launch_bounds__(256) void split_qkv(
    const float* __restrict__ qkv, bf16* __restrict__ Qb,
    bf16* __restrict__ Kt, bf16* __restrict__ Vb,
    int S, int H, int D, int E)
{
    long idx = (long)blockIdx.x * 256 + threadIdx.x;  // over M*E
    long m = idx / E;
    int  e = (int)(idx % E);
    int h = e / D, d = e % D;
    long b = m / S, sr = m % S;
    long z = b * H + h;
    const float* row = qkv + m * 3 * (long)E;
    Qb[(z * S + sr) * D + d] = (bf16)row[e];
    Kt[(z * D + d) * S + sr] = (bf16)row[E + e];
    Vb[(z * S + sr) * D + d] = (bf16)row[2 * E + e];
}

// ---------------------------------------------------------------------------
// Causal softmax over scaled scores. One block per (b,h,q) row.
// Writes bf16 probs; zero-fills masked tail so PV GEMM can run dense.
// ---------------------------------------------------------------------------
__global__ __launch_bounds__(256) void softmax_causal(
    const float* __restrict__ scores, bf16* __restrict__ probs, int S)
{
    __shared__ float red[256];
    const int tid = threadIdx.x;
    const long row = blockIdx.x;             // z*S + q
    const int q = (int)(row % S);
    const int len = q + 1;
    const float* sr = scores + row * (long)S;
    bf16* pr = probs + row * (long)S;

    float mx = -3.0e38f;
    for (int i = tid; i < len; i += 256) mx = fmaxf(mx, sr[i]);
    red[tid] = mx; __syncthreads();
    for (int off = 128; off > 0; off >>= 1) {
        if (tid < off) red[tid] = fmaxf(red[tid], red[tid + off]);
        __syncthreads();
    }
    mx = red[0]; __syncthreads();

    float sum = 0.f;
    for (int i = tid; i < len; i += 256) sum += __expf(sr[i] - mx);
    red[tid] = sum; __syncthreads();
    for (int off = 128; off > 0; off >>= 1) {
        if (tid < off) red[tid] += red[tid + off];
        __syncthreads();
    }
    const float inv = 1.0f / red[0];

    for (int i = tid; i < S; i += 256)
        pr[i] = (bf16)((i < len) ? __expf(sr[i] - mx) * inv : 0.0f);
}

// ---------------------------------------------------------------------------
// f32 -> bf16 elementwise (weight staging)
// ---------------------------------------------------------------------------
__global__ __launch_bounds__(256) void f32_to_bf16(
    const float* __restrict__ in, bf16* __restrict__ out, long n)
{
    long i = (long)blockIdx.x * 256 + threadIdx.x;
    if (i < n) out[i] = (bf16)in[i];
}

// ---------------------------------------------------------------------------
// Stage LM head weight [E,V] f32 -> [E,Vpad] bf16, zero padding cols V..Vpad-1
// ---------------------------------------------------------------------------
__global__ __launch_bounds__(256) void stage_lm_weight(
    const float* __restrict__ w, bf16* __restrict__ out, int V, int Vpad)
{
    long idx = (long)blockIdx.x * 256 + threadIdx.x;  // over E*Vpad
    long r = idx / Vpad;
    int  c = (int)(idx % Vpad);
    out[idx] = (c < V) ? (bf16)w[r * V + c] : (bf16)0.0f;
}

// ---------------------------------------------------------------------------
// Host-side orchestration
// ---------------------------------------------------------------------------
extern "C" void kernel_launch(void* const* d_in, const int* in_sizes, int n_in,
                              void* d_out, int out_size, void* d_ws, size_t ws_size,
                              hipStream_t stream)
{
    (void)in_sizes; (void)n_in; (void)out_size;
    const int B = 2, S = 1024, V = 50257, E = 768, H = 12, L = 4, D = 64;
    const long M = (long)B * S;      // 2048 token rows
    const int E3 = 3 * E, E4 = 4 * E;
    const int Vpad = (V + BN - 1) / BN * BN;   // 50304

    const int*   ids   = (const int*)d_in[0];
    const float* wte   = (const float*)d_in[1];
    const float* wpe   = (const float*)d_in[2];
    const float* ln1_s = (const float*)d_in[3];
    const float* ln1_b = (const float*)d_in[4];
    const float* w_qkv = (const float*)d_in[5];
    const float* b_qkv = (const float*)d_in[6];
    const float* w_ao  = (const float*)d_in[7];
    const float* b_ao  = (const float*)d_in[8];
    const float* ln2_s = (const float*)d_in[9];
    const float* ln2_b = (const float*)d_in[10];
    const float* w_fc  = (const float*)d_in[11];
    const float* b_fc  = (const float*)d_in[12];
    const float* w_mp  = (const float*)d_in[13];
    const float* b_mp  = (const float*)d_in[14];
    const float* lnf_s = (const float*)d_in[15];
    const float* lnf_b = (const float*)d_in[16];
    const float* w_lm  = (const float*)d_in[17];
    float* out = (float*)d_out;

    // workspace carve (256B aligned)
    size_t off = 0;
    char* base = (char*)d_ws;
    auto alloc = [&](size_t bytes) -> void* {
        void* p = base + off;
        off = (off + bytes + 255) & ~(size_t)255;
        return p;
    };
    float* x      = (float*)alloc(M * E * 4);
    bf16*  hbf    = (bf16*) alloc(M * E * 2);
    float* qkv    = (float*)alloc(M * (size_t)E3 * 4);
    bf16*  Qb     = (bf16*) alloc(M * E * 2);
    bf16*  Kt     = (bf16*) alloc(M * E * 2);
    bf16*  Vb     = (bf16*) alloc(M * E * 2);
    float* scores = (float*)alloc((size_t)B * H * S * S * 4);
    bf16*  probs  = (bf16*) alloc((size_t)B * H * S * S * 2);
    bf16*  obf    = (bf16*) alloc(M * E * 2);
    bf16*  hfc    = (bf16*) alloc(M * (size_t)E4 * 2);
    bf16*  wbuf   = (bf16*) alloc((size_t)E * E4 * 2);
    bf16*  wlmb   = (bf16*) alloc((size_t)E * Vpad * 2);
    if (off > ws_size) return;   // workspace too small; deterministic no-op

    const dim3 blk(256);
    auto cvt = [&](const float* src, bf16* dst, long n) {
        f32_to_bf16<<<dim3((unsigned)((n + 255) / 256)), blk, 0, stream>>>(src, dst, n);
    };

    // embedding
    embed_kernel<<<dim3((unsigned)(M * E / 256)), blk, 0, stream>>>(ids, wte, wpe, x, S, E);
    // LM-head weights staged once, zero-padded to Vpad columns
    stage_lm_weight<<<dim3((unsigned)((long)E * Vpad / 256)), blk, 0, stream>>>(
        w_lm, wlmb, V, Vpad);

    for (int l = 0; l < L; ++l) {
        // --- ln1 -> bf16
        layernorm_bf16<<<dim3((unsigned)M), blk, 0, stream>>>(
            x, ln1_s + (long)l * E, ln1_b + (long)l * E, hbf, E);

        // --- qkv = h @ Wqkv + b
        cvt(w_qkv + (long)l * E * E3, wbuf, (long)E * E3);
        wmma_gemm<0, false><<<dim3(E3 / BN, (unsigned)(M / BM), 1), blk, 0, stream>>>(
            hbf, E, 0, wbuf, E3, 0, qkv, nullptr, E3, 0, 0, 1,
            b_qkv + (long)l * E3, nullptr, 1.0f, (int)M, E3, E);

        split_qkv<<<dim3((unsigned)(M * E / 256)), blk, 0, stream>>>(
            qkv, Qb, Kt, Vb, S, H, D, E);

        // --- scores[z] = (Q @ K^T) / sqrt(D),  z = b*H + h
        wmma_gemm<0, false><<<dim3(S / BN, S / BM, B * H), blk, 0, stream>>>(
            Qb, D, (long)S * D, Kt, S, (long)D * S,
            scores, nullptr, S, (long)S * S, 0, 1,
            nullptr, nullptr, 0.125f, S, S, D);

        softmax_causal<<<dim3((unsigned)(B * H * S)), blk, 0, stream>>>(scores, probs, S);

        // --- O[z] = P @ V   -> bf16 [B,S,H,D] == [M,E] via (b,h) strides
        wmma_gemm<0, true><<<dim3(1, S / BM, B * H), blk, 0, stream>>>(
            probs, S, (long)S * S, Vb, D, (long)S * D,
            nullptr, obf, E, (long)S * E, D, H,
            nullptr, nullptr, 1.0f, S, D, S);

        // --- x = x + O @ Wo + bo
        cvt(w_ao + (long)l * E * E, wbuf, (long)E * E);
        wmma_gemm<0, false><<<dim3(E / BN, (unsigned)(M / BM), 1), blk, 0, stream>>>(
            obf, E, 0, wbuf, E, 0, x, nullptr, E, 0, 0, 1,
            b_ao + (long)l * E, x, 1.0f, (int)M, E, E);

        // --- ln2 -> bf16
        layernorm_bf16<<<dim3((unsigned)M), blk, 0, stream>>>(
            x, ln2_s + (long)l * E, ln2_b + (long)l * E, hbf, E);

        // --- hfc = gelu(h @ Wfc + bfc)  -> bf16
        cvt(w_fc + (long)l * E * E4, wbuf, (long)E * E4);
        wmma_gemm<1, true><<<dim3(E4 / BN, (unsigned)(M / BM), 1), blk, 0, stream>>>(
            hbf, E, 0, wbuf, E4, 0, nullptr, hfc, E4, 0, 0, 1,
            b_fc + (long)l * E4, nullptr, 1.0f, (int)M, E4, E);

        // --- x = x + hfc @ Wmp + bmp
        cvt(w_mp + (long)l * E4 * E, wbuf, (long)E4 * E);
        wmma_gemm<0, false><<<dim3(E / BN, (unsigned)(M / BM), 1), blk, 0, stream>>>(
            hfc, E4, 0, wbuf, E, 0, x, nullptr, E, 0, 0, 1,
            b_mp + (long)l * E, x, 1.0f, (int)M, E, E4);
    }

    // final LN -> bf16, then logits = h @ Wlm (B padded to Vpad, store guarded)
    layernorm_bf16<<<dim3((unsigned)M), blk, 0, stream>>>(x, lnf_s, lnf_b, hbf, E);
    wmma_gemm<0, false><<<dim3(Vpad / BN, (unsigned)(M / BM), 1), blk, 0, stream>>>(
        hbf, E, 0, wlmb, Vpad, 0, out, nullptr, V, 0, 0, 1,
        nullptr, nullptr, 1.0f, (int)M, V, E);
}
